// Encoder_20512763806236
// MI455X (gfx1250) — matrix operand order repaired and checked by reading, not compiled, not run on hardware
//
#include <hip/hip_runtime.h>
#include <hip/hip_bf16.h>

typedef __attribute__((ext_vector_type(2)))  float    v2f;
typedef __attribute__((ext_vector_type(8)))  float    v8f;
typedef __attribute__((ext_vector_type(8)))  _Float16 v8h;
typedef __attribute__((ext_vector_type(16))) _Float16 v16h;

union V16 { v16h v; v8h h[2]; };

#define B_    64
#define T_    2048
#define E_    300
#define H_    256
#define G4_   1024      // 4*H
#define BH_   (B_ * H_) // 16384
#define CHUNK 128       // time chunk; Z buffer = 64*128*1024 f32 = 32 MB

// -------------------------------------------------------------------------
// Prep: decompose W_hh (f32 [4H,H]) into f16 hi/lo planes for the split-f16
// recurrent GEMM:  w ~= (f32)whi + (f32)wlo,  |wlo| <= 2^-11 |w|.
// -------------------------------------------------------------------------
__global__ __launch_bounds__(256) void wsplit_kernel(
    const float* __restrict__ W, _Float16* __restrict__ Whi, _Float16* __restrict__ Wlo)
{
    const int i = blockIdx.x * 256 + threadIdx.x;     // 4H*H = 262144 total
    const float w = W[i];
    const _Float16 hi = (_Float16)w;
    Whi[i] = hi;
    Wlo[i] = (_Float16)(w - (float)hi);
}

// -------------------------------------------------------------------------
// Phase 1: Z[b*CHUNK+tc, n] = x[b, t0+tc, :] @ W_ih[n, :] + b_ih[n] + b_hh[n]
// Full-grid fp32 WMMA GEMM (exact f32): M = 8192, N = 1024, K = 300.
// -------------------------------------------------------------------------
__global__ __launch_bounds__(256) void xproj_kernel(
    const float* __restrict__ x,      // [B, T, E]
    const float* __restrict__ W_ih,   // [4H, E]
    const float* __restrict__ b_ih,   // [4H]
    const float* __restrict__ b_hh,   // [4H]
    float* __restrict__ Z,            // [B*CHUNK, 4H]
    int t0)
{
    const int lane  = threadIdx.x & 31;
    const int wave  = blockIdx.x * 8 + (threadIdx.x >> 5);
    const int nTile = wave & 63;          // 64 column tiles (4H/16)
    const int mTile = wave >> 6;          // 512 row tiles (B*CHUNK/16)
    const int lo = lane & 15;
    const int hi = lane >> 4;

    const int rrA = mTile * 16 + lo;              // row in [0, B*CHUNK)
    const int bA  = rrA >> 7;                     // / CHUNK
    const int tcA = rrA & (CHUNK - 1);
    const float* __restrict__ xrow = x + (size_t)(bA * T_ + t0 + tcA) * E_;

    const int nB = nTile * 16 + lo;
    const float* __restrict__ wrow = W_ih + (size_t)nB * E_;

    const float bias = b_ih[nB] + b_hh[nB];
    v8f c;
#pragma unroll
    for (int r = 0; r < 8; ++r) c[r] = bias;

#pragma unroll 4
    for (int k = 0; k < E_; k += 4) {
        v2f a  = *(const v2f*)(xrow + k + 2 * hi);
        v2f bv = *(const v2f*)(wrow + k + 2 * hi);
        c = __builtin_amdgcn_wmma_f32_16x16x4_f32(
                false, a, false, bv, (short)0, c, false, false);
    }

#pragma unroll
    for (int r = 0; r < 8; ++r) {
        int rr = mTile * 16 + r + 8 * hi;
        Z[(size_t)rr * G4_ + nB] = c[r];
    }
}

// -------------------------------------------------------------------------
// Phase 2: sequential skip-LSTM scan, one persistent 1024-thread workgroup.
//   h : LDS, f16 hi/lo planes, ping-pong buffers -> ONE barrier per step
//   c : f32 in registers (tile-owned per wave for the whole sequence)
//   recurrent GEMM: V_WMMA_F32_16X16X32_F16, 3-term hi/lo split (~f32 acc.)
// LDS layout (offsets in _Float16 units from hsm):
//   buffer b (b=0,1) base = b*2*BH_ ; [base, base+BH_) = hi, [base+BH_, +BH_) = lo
// Task = (mb, jb) 16x16 h-tile; its 4 gate tiles share the same A operand.
// Per wave/step: 2 tasks x 8 k-blocks x 4 gates x 3 terms = 192 WMMA (K=32).
// -------------------------------------------------------------------------
__global__ __launch_bounds__(1024) void scan_kernel(
    const float*    __restrict__ Z,        // [B*CHUNK, 4H] f32 gate preacts
    const _Float16* __restrict__ Whi,      // [4H, H] f16 hi plane
    const _Float16* __restrict__ Wlo,      // [4H, H] f16 lo plane
    const float*    __restrict__ mask,     // [B, T]
    const int*      __restrict__ lengths,  // [B]
    float* __restrict__ hstate,            // [B, H] f32 carry between chunks
    float* __restrict__ cstate,            // [B, H]
    float* __restrict__ lastH,             // [B, H] h at t == lengths-1
    int t0)
{
    extern __shared__ _Float16 hsm[];      // [2 bufs][hi|lo][BH_] = 128 KB

    const int tid  = threadIdx.x;
    const int lane = tid & 31;
    const int wave = tid >> 5;
    const int lo = lane & 15;
    const int hi = lane >> 4;

    // init h (buffer 0) as hi/lo f16 split
    for (int i = tid; i < BH_; i += 1024) {
        const float v = (t0 == 0) ? 0.0f : hstate[i];
        const _Float16 vh = (_Float16)v;
        hsm[i]       = vh;                           // buffer0 hi
        hsm[BH_ + i] = (_Float16)(v - (float)vh);    // buffer0 lo
    }

    v8f creg[2];
    int mbA[2], jbA[2];
    int idxv[2][8];
#pragma unroll
    for (int q = 0; q < 2; ++q) {
        const int task = wave * 2 + q;        // 0..63
        const int mb = task >> 4;             // 0..3  row tile (batch rows)
        const int jb = task & 15;             // 0..15 h-column tile
        mbA[q] = mb; jbA[q] = jb;
        const int jcol = jb * 16 + lo;
#pragma unroll
        for (int r = 0; r < 8; ++r) {
            const int brow = mb * 16 + r + 8 * hi;
            int ix = lengths[brow] - 1;
            idxv[q][r] = ix < 0 ? 0 : (ix > T_ - 1 ? T_ - 1 : ix);
            creg[q][r] = (t0 == 0) ? 0.0f : cstate[brow * H_ + jcol];
        }
    }
    __syncthreads();

    int p = 0;   // ping-pong phase
    for (int tc = 0; tc < CHUNK; ++tc) {
        const int t = t0 + tc;
        const int rdBase = p * (2 * BH_);            // read buffer base
        const int wrBase = (2 * BH_) - rdBase;       // write buffer base (1-p)
        const _Float16* __restrict__ hh = hsm + rdBase;         // old h hi
        const _Float16* __restrict__ hl = hsm + rdBase + BH_;   // old h lo
        _Float16* __restrict__ nh = hsm + wrBase;               // new h hi
        _Float16* __restrict__ nl = hsm + wrBase + BH_;         // new h lo

#pragma unroll
        for (int q = 0; q < 2; ++q) {
            const int mb = mbA[q], jb = jbA[q];
            const int jcol = jb * 16 + lo;

            if (tc + 1 < CHUNK)   // warm cache for next step's gate tile
                __builtin_prefetch(&Z[(size_t)((mb * 16) * CHUNK + tc + 1) * G4_ + jcol], 0, 1);

            // C init = f32 x-projection (+bias) from phase 1
            v8f zc[4];
#pragma unroll
            for (int gi = 0; gi < 4; ++gi) {
                const int n = gi * H_ + jcol;
#pragma unroll
                for (int r = 0; r < 8; ++r) {
                    const int brow = mb * 16 + r + 8 * hi;
                    zc[gi][r] = Z[(size_t)(brow * CHUNK + tc) * G4_ + n];
                }
            }

            // h @ W_hh^T, split-f16, f32 accumulate.
            // A(16x32) lane layout: K in [16hi,16hi+16), M = lane&15
            // B(32x16) lane layout: K in [16hi,16hi+16), N = lane&15
            const int arow = mb * 16 + lo;
            const _Float16* __restrict__ hrh = hh + arow * H_;
            const _Float16* __restrict__ hrl = hl + arow * H_;
#pragma unroll 2
            for (int k = 0; k < H_; k += 32) {
                V16 ah, al;
                ah.h[0] = *(const v8h*)(hrh + k + 16 * hi);
                ah.h[1] = *(const v8h*)(hrh + k + 16 * hi + 8);
                al.h[0] = *(const v8h*)(hrl + k + 16 * hi);
                al.h[1] = *(const v8h*)(hrl + k + 16 * hi + 8);
#pragma unroll
                for (int gi = 0; gi < 4; ++gi) {
                    const size_t wo = (size_t)(gi * H_ + jcol) * H_ + k + 16 * hi;
                    v16h bh = *(const v16h*)(Whi + wo);
                    v16h bl = *(const v16h*)(Wlo + wo);
                    zc[gi] = __builtin_amdgcn_wmma_f32_16x16x32_f16(
                                 false, ah.v, false, bh, (short)0, zc[gi], false, false);
                    zc[gi] = __builtin_amdgcn_wmma_f32_16x16x32_f16(
                                 false, ah.v, false, bl, (short)0, zc[gi], false, false);
                    zc[gi] = __builtin_amdgcn_wmma_f32_16x16x32_f16(
                                 false, al.v, false, bh, (short)0, zc[gi], false, false);
                }
            }

            // LSTM gates + skip-RNN mask select; write h2 into the OTHER buffer
#pragma unroll
            for (int r = 0; r < 8; ++r) {
                const int brow = mb * 16 + r + 8 * hi;
                const float ig = 1.0f / (1.0f + __expf(-zc[0][r]));
                const float fg = 1.0f / (1.0f + __expf(-zc[1][r]));
                const float gg = tanhf(zc[2][r]);
                const float og = 1.0f / (1.0f + __expf(-zc[3][r]));
                const float cn = fg * creg[q][r] + ig * gg;
                const float hn = og * tanhf(cn);
                const bool  rd = mask[brow * T_ + t] >= 0.5f;
                const int   hoff = brow * H_ + jcol;
                const float ho = (float)hh[hoff] + (float)hl[hoff];   // old h
                const float h2 = rd ? hn : ho;
                creg[q][r] = rd ? cn : creg[q][r];
                const _Float16 h2h = (_Float16)h2;
                nh[hoff] = h2h;
                nl[hoff] = (_Float16)(h2 - (float)h2h);
                if (idxv[q][r] == t) lastH[hoff] = h2;
            }
        }
        __syncthreads();   // new-h buffer visible; old buffer free to overwrite
        p ^= 1;
    }

    // persist carry for next chunk (CHUNK is even -> final state is buffer 0)
    const int fBase = p * (2 * BH_);
    for (int i = tid; i < BH_; i += 1024)
        hstate[i] = (float)hsm[fBase + i] + (float)hsm[fBase + BH_ + i];
#pragma unroll
    for (int q = 0; q < 2; ++q) {
        const int mb = mbA[q], jb = jbA[q];
        const int jcol = jb * 16 + lo;
#pragma unroll
        for (int r = 0; r < 8; ++r) {
            const int brow = mb * 16 + r + 8 * hi;
            cstate[brow * H_ + jcol] = creg[q][r];
        }
    }
}

// -------------------------------------------------------------------------
// Phase 3: 2-class head + softmax + argmax
// -------------------------------------------------------------------------
__global__ void head_kernel(const float* __restrict__ lastH,
                            const float* __restrict__ W_out,  // [2, H]
                            const float* __restrict__ b_out,  // [2]
                            float* __restrict__ out)          // [64*2 probs | 64 preds]
{
    const int b = threadIdx.x;
    if (b >= B_) return;
    float a0 = b_out[0], a1 = b_out[1];
    for (int k = 0; k < H_; ++k) {
        const float h = lastH[b * H_ + k];
        a0 += h * W_out[k];
        a1 += h * W_out[H_ + k];
    }
    const float m  = fmaxf(a0, a1);
    const float e0 = __expf(a0 - m), e1 = __expf(a1 - m);
    const float s  = e0 + e1;
    out[b * 2 + 0]  = e0 / s;
    out[b * 2 + 1]  = e1 / s;
    out[2 * B_ + b] = (e1 > e0) ? 1.0f : 0.0f;   // argmax (ties -> 0)
}

extern "C" void kernel_launch(void* const* d_in, const int* in_sizes, int n_in,
                              void* d_out, int out_size, void* d_ws, size_t ws_size,
                              hipStream_t stream) {
    const float* embedded = (const float*)d_in[0];  // [B,T,E]
    const float* mask     = (const float*)d_in[1];  // [B,T]
    const int*   lengths  = (const int*)  d_in[2];  // [B]
    const float* W_ih     = (const float*)d_in[3];  // [4H,E]
    const float* W_hh     = (const float*)d_in[4];  // [4H,H]
    const float* b_ih     = (const float*)d_in[5];  // [4H]
    const float* b_hh     = (const float*)d_in[6];  // [4H]
    const float* W_out    = (const float*)d_in[7];  // [2,H]
    const float* b_out    = (const float*)d_in[8];  // [2]
    float* out = (float*)d_out;

    float* ws     = (float*)d_ws;
    float* Z      = ws;                                   // 64*128*1024 f32 (32 MB)
    float* hstate = Z + (size_t)B_ * CHUNK * G4_;
    float* cstate = hstate + BH_;
    float* lastH  = cstate + BH_;
    _Float16* Whi = (_Float16*)(lastH + BH_);             // 4H*H f16
    _Float16* Wlo = Whi + (size_t)G4_ * H_;

    const size_t scanLds = 4ull * BH_ * sizeof(_Float16);       // 128 KB
    (void)hipFuncSetAttribute((const void*)scan_kernel,
                              hipFuncAttributeMaxDynamicSharedMemorySize,
                              (int)scanLds);

    wsplit_kernel<<<(G4_ * H_) / 256, 256, 0, stream>>>(W_hh, Whi, Wlo);

    const int rowTiles = (B_ * CHUNK) / 16;               // 512
    const int blocks   = rowTiles * (G4_ / 16) / 8;       // 4096 blocks x 8 waves

    for (int t0 = 0; t0 < T_; t0 += CHUNK) {
        xproj_kernel<<<blocks, 256, 0, stream>>>(embedded, W_ih, b_ih, b_hh, Z, t0);
        scan_kernel<<<1, 1024, (unsigned)scanLds, stream>>>(
            Z, Whi, Wlo, mask, lengths, hstate, cstate, lastH, t0);
    }
    head_kernel<<<1, 64, 0, stream>>>(lastH, W_out, b_out, out);
}
